// AttentionTSSA_11802570129741
// MI455X (gfx1250) — compile-verified, matrix operand order checked by reading, not run on hardware
//
#include <hip/hip_runtime.h>

// ---------------- types ----------------
typedef __bf16 bf16;
typedef __attribute__((ext_vector_type(16))) __bf16 v16bf;
typedef __attribute__((ext_vector_type(8)))  __bf16 v8bf;
typedef __attribute__((ext_vector_type(8)))  float  v8f;

#define Bb   8
#define Nn   4096
#define DIMM 1024
#define Hh   16
#define Dd   64
#define BH   (Bb * Hh)          // 128
#define MROWS (Bb * Nn)         // 32768

__device__ __forceinline__ bf16 to_bf16(float f) {
    unsigned u = __builtin_bit_cast(unsigned, f);
    unsigned r = u + 0x7FFFu + ((u >> 16) & 1u);   // round-to-nearest-even
    unsigned short h = (unsigned short)(r >> 16);
    return __builtin_bit_cast(bf16, h);
}

// Load one 16-bit WMMA fragment (CDNA5 lane layout):
// lanes 0-15 take K = [0..7] and [16..23]; lanes 16-31 take K = [8..15] and [24..31]
// (the +kb per-lane base is folded into the pointer by the caller).
__device__ __forceinline__ v16bf load_frag(const bf16* __restrict__ p) {
    v8bf lo = *(const v8bf*)p;          // 16B
    v8bf hi = *(const v8bf*)(p + 16);   // 16B
    return __builtin_shufflevector(lo, hi, 0,1,2,3,4,5,6,7,8,9,10,11,12,13,14,15);
}

// ---------------- fp32 -> bf16 convert ----------------
__global__ void cvt_bf16_kernel(const float* __restrict__ in, bf16* __restrict__ out, long n) {
    long i = (long)blockIdx.x * blockDim.x + threadIdx.x;
    long stride = (long)gridDim.x * blockDim.x;
    for (; i < n; i += stride) out[i] = to_bf16(in[i]);
}

// ---------------- WMMA bf16 NT GEMM, 32x64 macro-tile per wave ----------------
// C[m,j] = sum_k A[m,k] * Bm[j,k]   (both K-contiguous, K % 32 == 0)
// mode 0: scatter into w[(b*H+h)*N*D + n*D + d]  with m=(b,n), j=(h,d)
// mode 1: Cout[m*N + j] = acc + bias[j]
// __launch_bounds__(128, 1): WMMA waves don't need occupancy; give the register
// allocator the full VGPR file so accumulators + double-buffered fragments stay
// pinned in even-aligned tuples (no v_mov_b64 shuffling / hazard v_nops).
__global__ void __launch_bounds__(128, 1)
gemm_nt_wmma(const bf16* __restrict__ A, const bf16* __restrict__ Bm,
             const float* __restrict__ bias, float* __restrict__ Cout,
             int N, int K, int mode) {
    const int wave = threadIdx.x >> 5;
    const int lane = threadIdx.x & 31;
    const int tileM = blockIdx.y * 32;                 // 2 M-subtiles of 16
    const int tileN = (blockIdx.x * 4 + wave) * 64;    // 4 N-subtiles of 16

    const int ml = lane & 15;                 // row-in-subtile (A) / col-in-subtile (B)
    const int kb = (lane & 16) ? 8 : 0;       // per-lane K sub-block base

    const bf16* arow[2];
    const bf16* brow[4];
#pragma unroll
    for (int i = 0; i < 2; ++i) arow[i] = A  + (size_t)(tileM + 16 * i + ml) * K + kb;
#pragma unroll
    for (int j = 0; j < 4; ++j) brow[j] = Bm + (size_t)(tileN + 16 * j + ml) * K + kb;

    v8f acc[2][4] = {};

    for (int k0 = 0; k0 < K; k0 += 32) {
        v16bf af[2], bfv[4];
#pragma unroll
        for (int i = 0; i < 2; ++i) af[i]  = load_frag(arow[i] + k0);
#pragma unroll
        for (int j = 0; j < 4; ++j) bfv[j] = load_frag(brow[j] + k0);
#pragma unroll
        for (int i = 0; i < 2; ++i)
#pragma unroll
            for (int j = 0; j < 4; ++j)
                acc[i][j] = __builtin_amdgcn_wmma_f32_16x16x32_bf16(
                    /*neg_a=*/false, af[i], /*neg_b=*/false, bfv[j],
                    /*c_mod=*/(short)0, acc[i][j], /*reuse_a=*/false, /*reuse_b=*/false);
    }

    // C/D layout: VGPR r, lanes 0-15 -> M=r, N=lane; lanes 16-31 -> M=r+8, N=lane-16
    const int mo = (lane & 16) ? 8 : 0;
    const int nl = lane & 15;
#pragma unroll
    for (int i = 0; i < 2; ++i) {
#pragma unroll
        for (int j = 0; j < 4; ++j) {
            const int jj = tileN + 16 * j + nl;
#pragma unroll
            for (int r = 0; r < 8; ++r) {
                const int m = tileM + 16 * i + r + mo;
                const float v = acc[i][j][r];
                if (mode == 0) {
                    const int b = m >> 12, n = m & (Nn - 1);
                    const int h = jj >> 6, d = jj & (Dd - 1);
                    Cout[(size_t)(b * Hh + h) * (Nn * Dd) + (size_t)n * Dd + d] = v;
                } else {
                    Cout[(size_t)m * N + jj] = v + bias[jj];
                }
            }
        }
    }
}

// ---------------- inv2[bh][d] = 1 / max(||w[:,d]||, eps)^2 ----------------
__global__ void norm_kernel(const float* __restrict__ w, float* __restrict__ inv2) {
    const int bh = blockIdx.x;
    const int t  = threadIdx.x;          // 256
    const int d  = t & 63, g = t >> 6;   // 4 n-groups x 64 d
    const float* base = w + (size_t)bh * Nn * Dd;
    float acc = 0.f;
    for (int n = g; n < Nn; n += 4) {
        float v = base[(size_t)n * Dd + d];
        acc += v * v;
    }
    __shared__ float red[256];
    red[t] = acc;
    __syncthreads();
    if (t < 64) {
        float s = red[t] + red[t + 64] + red[t + 128] + red[t + 192];
        float nrm = sqrtf(s);
        float den = fmaxf(nrm, 1e-12f);
        inv2[bh * Dd + t] = 1.f / (den * den);
    }
}

// ---------------- ss[bh][n] = sum_d w^2 * inv2[d] ----------------
__global__ void ss_kernel(const float* __restrict__ w, const float* __restrict__ inv2,
                          float* __restrict__ ss) {
    const int bh = blockIdx.y;
    const int t  = threadIdx.x;          // 256
    const int d  = t & 63, nl = t >> 6;
    __shared__ float sinv[64];
    __shared__ float red[256];
    if (t < 64) sinv[t] = inv2[bh * Dd + t];
    __syncthreads();
    const float* base = w + (size_t)bh * Nn * Dd;
    for (int c = 0; c < 32; ++c) {
        const int n = blockIdx.x * 128 + c * 4 + nl;
        float v = base[(size_t)n * Dd + d];
        red[t] = v * v * sinv[d];
        __syncthreads();
        for (int st = 32; st > 0; st >>= 1) {
            if (d < st) red[t] += red[t + st];
            __syncthreads();
        }
        if (d == 0) ss[(size_t)bh * Nn + n] = red[nl * 64];
        __syncthreads();
    }
}

// ---------------- zero small buffer ----------------
__global__ void zero_kernel(float* p, int n) {
    int i = blockIdx.x * blockDim.x + threadIdx.x;
    if (i < n) p[i] = 0.f;
}

// ---------------- softmax over heads + partial sumPi ----------------
__global__ void softmax_heads_kernel(const float* __restrict__ ss, const float* __restrict__ temp,
                                     float* __restrict__ Pi, float* __restrict__ sumPi) {
    const int idx = blockIdx.x * blockDim.x + threadIdx.x;   // (b,n) flat, 32768
    const int b = idx >> 12, n = idx & (Nn - 1);
    __shared__ float part[16];
    if (threadIdx.x < 16) part[threadIdx.x] = 0.f;
    __syncthreads();

    float l[16];
    float mx = -3.4e38f;
#pragma unroll
    for (int h = 0; h < Hh; ++h) {
        l[h] = ss[(size_t)(b * Hh + h) * Nn + n] * temp[h];
        mx = fmaxf(mx, l[h]);
    }
    float s = 0.f;
#pragma unroll
    for (int h = 0; h < Hh; ++h) { l[h] = __expf(l[h] - mx); s += l[h]; }
    const float inv = 1.f / s;
#pragma unroll
    for (int h = 0; h < Hh; ++h) {
        float p = l[h] * inv;
        Pi[(size_t)(b * Hh + h) * Nn + n] = p;
        atomicAdd(&part[h], p);
    }
    __syncthreads();
    if (threadIdx.x < 16) atomicAdd(&sumPi[b * Hh + threadIdx.x], part[threadIdx.x]);
}

// ---------------- dots + attn = 1/(1+dots) ----------------
__global__ void dots_attn_kernel(const float* __restrict__ w, const float* __restrict__ Pi,
                                 const float* __restrict__ sumPi, float* __restrict__ attn) {
    const int bh = blockIdx.x;
    const int t  = threadIdx.x;
    const int d  = t & 63, g = t >> 6;
    const float* base = w + (size_t)bh * Nn * Dd;
    const float* pib  = Pi + (size_t)bh * Nn;
    float acc = 0.f;
    for (int n = g; n < Nn; n += 4) {
        float v = base[(size_t)n * Dd + d];
        acc += pib[n] * v * v;
    }
    __shared__ float red[256];
    red[t] = acc;
    __syncthreads();
    if (t < 64) {
        float s = red[t] + red[t + 64] + red[t + 128] + red[t + 192];
        float dots = s / (sumPi[bh] + 1e-8f);
        attn[bh * Dd + t] = 1.f / (1.f + dots);
    }
}

// ---------------- mid[b,n,h*64+d] = bf16( -(w*Pi)*attn ) ----------------
__global__ void mid_kernel(const float* __restrict__ w, const float* __restrict__ Pi,
                           const float* __restrict__ attn, bf16* __restrict__ mid) {
    long i = (long)blockIdx.x * blockDim.x + threadIdx.x;
    long stride = (long)gridDim.x * blockDim.x;
    const long total = (long)BH * Nn * Dd;
    for (; i < total; i += stride) {
        const int d  = (int)(i & 63);
        const long r = i >> 6;
        const int n  = (int)(r & (Nn - 1));
        const int bh = (int)(r >> 12);
        const int b = bh >> 4, h = bh & 15;
        float v = -(w[i] * Pi[(size_t)bh * Nn + n]) * attn[bh * Dd + d];
        mid[((size_t)(b * Nn + n)) * DIMM + h * Dd + d] = to_bf16(v);
    }
}

// ---------------- host-side orchestration ----------------
extern "C" void kernel_launch(void* const* d_in, const int* in_sizes, int n_in,
                              void* d_out, int out_size, void* d_ws, size_t ws_size,
                              hipStream_t stream) {
    const float* x    = (const float*)d_in[0];   // (8,4096,1024)
    const float* Wqkv = (const float*)d_in[1];   // (1024,1024)
    const float* temp = (const float*)d_in[2];   // (16,1)
    const float* Wout = (const float*)d_in[3];   // (1024,1024)
    const float* bout = (const float*)d_in[4];   // (1024,)
    float* out = (float*)d_out;

    char* ws = (char*)d_ws;
    size_t off = 0;
    float* w_f32   = (float*)(ws + off); off += (size_t)MROWS * DIMM * 4;   // 128 MB
    bf16*  xm_bf16 = (bf16*) (ws + off); off += (size_t)MROWS * DIMM * 2;   // 64 MB (x, then mid)
    bf16*  Wq_bf16 = (bf16*) (ws + off); off += (size_t)DIMM * DIMM * 2;    // 2 MB
    bf16*  Wo_bf16 = (bf16*) (ws + off); off += (size_t)DIMM * DIMM * 2;    // 2 MB
    float* Pi      = (float*)(ws + off); off += (size_t)BH * Nn * 4;        // 2 MB
    float* ssb     = (float*)(ws + off); off += (size_t)BH * Nn * 4;        // 2 MB
    float* inv2    = (float*)(ws + off); off += (size_t)BH * Dd * 4;
    float* attn    = (float*)(ws + off); off += (size_t)BH * Dd * 4;
    float* sumPi   = (float*)(ws + off); off += 512;
    (void)ws_size; (void)in_sizes; (void)n_in; (void)out_size;

    // 1) convert inputs to bf16
    cvt_bf16_kernel<<<4096, 256, 0, stream>>>(x,    xm_bf16, (long)MROWS * DIMM);
    cvt_bf16_kernel<<<1024, 256, 0, stream>>>(Wqkv, Wq_bf16, (long)DIMM * DIMM);
    cvt_bf16_kernel<<<1024, 256, 0, stream>>>(Wout, Wo_bf16, (long)DIMM * DIMM);

    // 2) GEMM1: w = x @ Wqkv.T, scattered to (b,h,n,d)
    {
        dim3 grid(DIMM / 64 / 4, MROWS / 32);   // (4, 1024)
        gemm_nt_wmma<<<grid, 128, 0, stream>>>(xm_bf16, Wq_bf16, nullptr, w_f32,
                                               DIMM, DIMM, /*mode=*/0);
    }

    // 3) statistics chain
    norm_kernel<<<BH, 256, 0, stream>>>(w_f32, inv2);
    {
        dim3 grid(Nn / 128, BH);                // (32, 128)
        ss_kernel<<<grid, 256, 0, stream>>>(w_f32, inv2, ssb);
    }
    zero_kernel<<<1, 128, 0, stream>>>(sumPi, BH);
    softmax_heads_kernel<<<(Bb * Nn) / 256, 256, 0, stream>>>(ssb, temp, Pi, sumPi);
    dots_attn_kernel<<<BH, 256, 0, stream>>>(w_f32, Pi, sumPi, attn);

    // 4) mid = -(w * Pi) * attn   (bf16, reuses x_bf16 buffer, (b,n,dim) layout)
    mid_kernel<<<4096, 256, 0, stream>>>(w_f32, Pi, attn, xm_bf16);

    // 5) GEMM2: out = mid @ Wout.T + bout
    {
        dim3 grid(DIMM / 64 / 4, MROWS / 32);
        gemm_nt_wmma<<<grid, 128, 0, stream>>>(xm_bf16, Wo_bf16, bout, out,
                                               DIMM, DIMM, /*mode=*/1);
    }
}
